// CIN_52553219834054
// MI455X (gfx1250) — matrix operand order repaired and checked by reading, not compiled, not run on hardware
//
#include <hip/hip_runtime.h>

typedef __attribute__((ext_vector_type(16))) __bf16 v16bf;
typedef __attribute__((ext_vector_type(8)))  float  v8f;

// ---------------------------------------------------------------------------
// Prep kernel: convert W1/W2 (f32, 64 x 4096) to bf16 in d_ws, pre-swizzled
// into a WMMA A-operand lane layout (contiguous k order) so the main kernel's A
// loads are contiguous b128s.
//
// dest flat layout (bf16 elems): [l][i(64)][j0i(2)][ot(4)][lane(32)][e(16)]
//   A tile = rows o = ot*16 + lane%16, K-chunk k = j0i*32 + k_local
//   k_local = e + (lane>=16 ? 16 : 0)   (contiguous k order, matches B)
//   source column c = i*64 + j0i*32 + k_local  (c = i*F + j of the CIN einsum)
// ---------------------------------------------------------------------------
__global__ __launch_bounds__(256) void cin_prep_kernel(
    const float* __restrict__ W1, const float* __restrict__ W2,
    __bf16* __restrict__ wsw)
{
  int idx = blockIdx.x * 256 + threadIdx.x;           // 0 .. 524287
  int e    =  idx        & 15;
  int lane = (idx >> 4)  & 31;
  int ot   = (idx >> 9)  & 3;
  int j0i  = (idx >> 11) & 1;
  int i    = (idx >> 12) & 63;
  int l    = (idx >> 18) & 1;
  int hi   = lane >> 4;
  int klocal = e + hi * 16;
  int c = i * 64 + j0i * 32 + klocal;
  int o = ot * 16 + (lane & 15);
  const float* W = l ? W2 : W1;
  wsw[idx] = (__bf16)W[o * 4096 + c];
}

// ---------------------------------------------------------------------------
// Main kernel: 4 waves / block, one batch element per wave.
// Each wave computes its full 64x64 layer output as 16 WMMA f32 accumulators.
// Per K-step i, ALL 8 A tiles and ALL 8 B tiles (Z chunks, formed in
// registers via packed bf16 muls) are materialized first, then 32 WMMAs issue
// back-to-back: distinct operand registers kill the WMMA->VALU WAR v_nops.
// ---------------------------------------------------------------------------
__global__ __launch_bounds__(128) void cin_main_kernel(
    const float* __restrict__ x, const __bf16* __restrict__ wsw,
    const float* __restrict__ b1, const float* __restrict__ b2,
    const float* __restrict__ wfc, const float* __restrict__ bfc,
    float* __restrict__ out)
{
  // hx[0][w] = transposed x (xt[d][i]),  hx[1][w] = transposed h1 after layer 1
  __shared__ alignas(32) __bf16 hx[2][4][64 * 64];      // 64 KB
  __shared__ alignas(32) __bf16 Abuf[2][4096];          // 16 KB, double buffer
  __shared__ float wfc_s[128];
  __shared__ float bias_s[2][64];
  __shared__ float red[4];

  const int t      = threadIdx.x;
  const int w      = t >> 5;
  const int lane   = t & 31;
  const int lanelo = lane & 15;
  const int hi     = lane >> 4;
  const int b      = blockIdx.x * 4 + w;

  // small params -> LDS
  wfc_s[t] = wfc[t];
  if (t < 64) bias_s[0][t] = b1[t];
  else        bias_s[1][t - 64] = b2[t - 64];
  if (lane == 0) red[w] = 0.f;

  // load + transpose x[b] (f32 row-major [i][d]) -> hx[0][w] as bf16 [d][i]
  {
    const float4* xb = (const float4*)(x + (size_t)b * 4096);
    #pragma unroll 4
    for (int e = 0; e < 32; e++) {
      int idx = e * 32 + lane;            // float4 index over 4096 floats
      float4 v = xb[idx];
      int i  = idx >> 4;                  // 16 float4 per 64-wide row
      int d0 = (idx & 15) * 4;
      hx[0][w][(d0 + 0) * 64 + i] = (__bf16)v.x;
      hx[0][w][(d0 + 1) * 64 + i] = (__bf16)v.y;
      hx[0][w][(d0 + 2) * 64 + i] = (__bf16)v.z;
      hx[0][w][(d0 + 3) * 64 + i] = (__bf16)v.w;
    }
  }

  float lanep = 0.f;   // fused pooled * Wfc partial (both layers)

  for (int l = 0; l < 2; l++) {
    const __bf16* ht = &hx[l][w][0];                   // layer input, [d][j]
    const __bf16* wl = wsw + (size_t)l * 64 * 4096;    // swizzled W, 8KB per i

    v8f acc[16];
    #pragma unroll
    for (int q = 0; q < 16; q++)
      #pragma unroll
      for (int r = 0; r < 8; r++) acc[q][r] = 0.f;

    // cooperative stage of 8KB W slab for i -> Abuf[buf]
    auto stage = [&](int buf, int i) {
      const uint4* src = (const uint4*)(wl + i * 4096);
      uint4* dst = (uint4*)&Abuf[buf][0];
      #pragma unroll
      for (int q = 0; q < 4; q++) dst[t * 4 + q] = src[t * 4 + q];
    };

    stage(0, 0);
    __syncthreads();

    for (int i = 0; i < 64; i++) {
      const int buf = i & 1;
      if (i + 1 < 64) stage(buf ^ 1, i + 1);

      // --- materialize ALL operands for this K=64 slab first ---------------
      // A tiles: [j0i*4 + ot]
      v16bf Aa[8];
      #pragma unroll
      for (int j0i = 0; j0i < 2; j0i++)
        #pragma unroll
        for (int ot = 0; ot < 4; ot++)
          Aa[j0i * 4 + ot] =
              *(const v16bf*)(&Abuf[buf][j0i * 2048 + ot * 512 + lane * 16]);

      // x splats: lane's d = nt*16 + lanelo
      v16bf xv[4];
      #pragma unroll
      for (int nt = 0; nt < 4; nt++) {
        __bf16 s = hx[0][w][(nt * 16 + lanelo) * 64 + i];
        #pragma unroll
        for (int q = 0; q < 16; q++) xv[nt][q] = s;
      }

      // B tiles: Z_i[j,d] = x[i,d] * h[j,d], lane holds K = j0i*32 + hi*16 + q
      v16bf bz[8];                        // [j0i*4 + nt]
      #pragma unroll
      for (int j0i = 0; j0i < 2; j0i++)
        #pragma unroll
        for (int nt = 0; nt < 4; nt++) {
          const v16bf hv =
              *(const v16bf*)(&ht[(nt * 16 + lanelo) * 64 + j0i * 32 + hi * 16]);
          bz[j0i * 4 + nt] = hv * xv[nt];   // v_pk_mul_bf16 (op_sel splat)
        }

      // --- 32 back-to-back WMMAs -------------------------------------------
      #pragma unroll
      for (int j0i = 0; j0i < 2; j0i++)
        #pragma unroll
        for (int nt = 0; nt < 4; nt++)
          #pragma unroll
          for (int ot = 0; ot < 4; ot++)
            acc[ot * 4 + nt] = __builtin_amdgcn_wmma_f32_16x16x32_bf16(
                false, Aa[j0i * 4 + ot], false, bz[j0i * 4 + nt], (short)0,
                acc[ot * 4 + nt], false, false);

      __syncthreads();
    }

    // epilogue: bias + relu, fuse pooled*Wfc, store transposed h1 for layer 2
    #pragma unroll
    for (int ot = 0; ot < 4; ot++) {
      #pragma unroll
      for (int nt = 0; nt < 4; nt++) {
        const int d = nt * 16 + lanelo;
        #pragma unroll
        for (int r = 0; r < 8; r++) {
          const int o = ot * 16 + r + 8 * hi;
          float v = acc[ot * 4 + nt][r] + bias_s[l][o];
          v = fmaxf(v, 0.f);
          lanep += wfc_s[l * 64 + o] * v;
          if (l == 0) hx[1][w][d * 64 + o] = (__bf16)v;
        }
      }
    }
  }

  atomicAdd(&red[w], lanep);        // ds_add_f32, per-wave slot
  __syncthreads();
  if (lane == 0) out[b] = red[w] + bfc[0];
}

extern "C" void kernel_launch(void* const* d_in, const int* in_sizes, int n_in,
                              void* d_out, int out_size, void* d_ws, size_t ws_size,
                              hipStream_t stream) {
  (void)n_in; (void)out_size; (void)ws_size;
  const float* x   = (const float*)d_in[0];
  const float* W1  = (const float*)d_in[1];
  const float* b1  = (const float*)d_in[2];
  const float* W2  = (const float*)d_in[3];
  const float* b2  = (const float*)d_in[4];
  const float* wfc = (const float*)d_in[5];
  const float* bfc = (const float*)d_in[6];
  float* out = (float*)d_out;
  __bf16* wsw = (__bf16*)d_ws;          // needs 1 MB of scratch

  const int batches = in_sizes[0] / 4096;   // B = 1024

  hipLaunchKernelGGL(cin_prep_kernel, dim3(2048), dim3(256), 0, stream,
                     W1, W2, wsw);
  hipLaunchKernelGGL(cin_main_kernel, dim3(batches / 4), dim3(128), 0, stream,
                     x, wsw, b1, b2, wfc, bfc, out);
}
